// R2LTransformer_56178172231785
// MI455X (gfx1250) — compile-verified, matrix operand order checked by reading, not run on hardware
//
#include <hip/hip_runtime.h>
#include <hip/hip_bf16.h>

// ---------------------------------------------------------------------------
// R2L transformer for MI455X (gfx1250, wave32, WMMA).
// All GEMMs + attention cores run on v_wmma_f32_16x16x32_f16 (f32 accum).
// Rel-pos bias is fed through the WMMA C operand (q pre-scaled by 32^-0.5
// in the QKV projection epilogue), so the score loop has no LDS RMW chain.
// ---------------------------------------------------------------------------

typedef __attribute__((ext_vector_type(16))) _Float16 v16h;
typedef __attribute__((ext_vector_type(8)))  float    v8f;

constexpr int DIMM   = 256;
constexpr int HEADS  = 8;
constexpr int DH     = 32;
constexpr int NWIN   = 1024;   // 16 batches * 64 windows
constexpr int NTOK   = 65;     // 1 region token + 8x8 local
constexpr int NPAD   = 80;     // 65 padded to 5*16
constexpr int KPAD   = 96;     // attn K padded to 3*32
constexpr int RWIN   = 16;     // region attention "windows" (batches)
constexpr int RTOK   = 64;     // region tokens per batch
constexpr float EPSV  = 1e-5f;
constexpr float QSCALE = 0.17677669529663687f; // 32^-0.5

// ---------------- WMMA helpers ----------------

__device__ __forceinline__ v8f wmma_f16(v16h a, v16h b, v8f c) {
  // (neg_a, A, neg_b, B, c_mod, C, reuse_a, reuse_b)
  return __builtin_amdgcn_wmma_f32_16x16x32_f16(false, a, false, b, (short)0, c,
                                                false, false);
}

// Load a 16x32 f16 operand fragment from a row-major LDS tile.
// rowBase points at element [row][0] of a row with >=32 f16, 16B aligned.
// Lanes 0-15 hold K 0..7 / 16..23, lanes 16-31 hold K 8..15 / 24..31.
__device__ __forceinline__ v16h ldsFrag(const _Float16* rowBase, int half) {
  union { uint4 q[2]; v16h h; } u;
  u.q[0] = *(const uint4*)(rowBase + half * 8);
  u.q[1] = *(const uint4*)(rowBase + 16 + half * 8);
  return u.h;
}

__device__ __forceinline__ float geluf(float x) {
  float x3 = x * x * x;
  return 0.5f * x * (1.0f + tanhf(0.7978845608028654f * (x + 0.044715f * x3)));
}

// ---------------- Generic WMMA GEMM ----------------
// C[M,N] = A[M,K](f16) * Bt[N,K](f16)^T, epilogues:
//   EPI 0: store f16, scaling columns n<256 by QSCALE (QKV projection:
//          pre-scales q so attention scores need no separate scaling)
//   EPI 1: +bias, GELU, store f16
//   EPI 2: +bias, residual-add into f32 buffer
// Block tile 128x64, 8 waves (4x2), each wave 32x32.
// REQUIRES: M multiple of 128, N multiple of 64, K multiple of 32.

template <int EPI>
__global__ __launch_bounds__(256) void k_gemm(const _Float16* __restrict__ A,
                                              const _Float16* __restrict__ Bt,
                                              const float* __restrict__ bias,
                                              float* __restrict__ resid,
                                              _Float16* __restrict__ out,
                                              int M, int N, int K) {
  __shared__ _Float16 As[128 * 32];
  __shared__ _Float16 Bs[64 * 32];

  const int tid  = threadIdx.x;
  const int m0   = blockIdx.x * 128;
  const int n0   = blockIdx.y * 64;
  const int wave = tid >> 5;
  const int lane = tid & 31;
  const int wm   = wave & 3;   // 4 waves along M
  const int wn   = wave >> 2;  // 2 waves along N
  const int half = lane >> 4;
  const int lrow = lane & 15;

  v8f acc[2][2] = {};

  for (int k0 = 0; k0 < K; k0 += 32) {
    // stage A tile: 128 rows x 32 f16 = 512 16B-chunks
#pragma unroll
    for (int i = 0; i < 2; ++i) {
      int cid = tid + i * 256;
      int r = cid >> 2, c = cid & 3;
      *(uint4*)(&As[r * 32 + c * 8]) =
          *(const uint4*)(A + (size_t)(m0 + r) * K + k0 + c * 8);
    }
    // stage B tile: 64 rows x 32 f16 = 256 16B-chunks
    {
      int r = tid >> 2, c = tid & 3;
      *(uint4*)(&Bs[r * 32 + c * 8]) =
          *(const uint4*)(Bt + (size_t)(n0 + r) * K + k0 + c * 8);
    }
    __syncthreads();

    v16h a0 = ldsFrag(&As[(wm * 32 + lrow) * 32], half);
    v16h a1 = ldsFrag(&As[(wm * 32 + 16 + lrow) * 32], half);
    v16h b0 = ldsFrag(&Bs[(wn * 32 + lrow) * 32], half);
    v16h b1 = ldsFrag(&Bs[(wn * 32 + 16 + lrow) * 32], half);

    acc[0][0] = wmma_f16(a0, b0, acc[0][0]);
    acc[0][1] = wmma_f16(a0, b1, acc[0][1]);
    acc[1][0] = wmma_f16(a1, b0, acc[1][0]);
    acc[1][1] = wmma_f16(a1, b1, acc[1][1]);
    __syncthreads();
  }

#pragma unroll
  for (int i = 0; i < 2; ++i) {
#pragma unroll
    for (int j = 0; j < 2; ++j) {
      int n = n0 + wn * 32 + j * 16 + lrow;
#pragma unroll
      for (int r = 0; r < 8; ++r) {
        int m = m0 + wm * 32 + i * 16 + r + half * 8;
        float val = acc[i][j][r];
        if (EPI == 0) {
          float sc = (n < DIMM) ? QSCALE : 1.0f;  // scale the q block
          out[(size_t)m * N + n] = (_Float16)(val * sc);
        } else if (EPI == 1) {
          out[(size_t)m * N + n] = (_Float16)geluf(val + bias[n]);
        } else {
          resid[(size_t)m * N + n] += val + bias[n];
        }
      }
    }
  }
}

// ---------------- Fused per-(window, head) attention ----------------
// QKV: [nWin*nTok, 768] f16, q already scaled by 32^-0.5
// bias: (HEADS, nTok, nTok) f32 or nullptr
// O:   [nWin*nTok, 256] f16
// nPad multiple of 16 (>= nTok), kPad multiple of 32 (>= nPad).

__global__ __launch_bounds__(256) void k_attn(const _Float16* __restrict__ QKV,
                                              const float* __restrict__ bias,
                                              _Float16* __restrict__ O,
                                              int nTok, int nPad, int kPad) {
  __shared__ _Float16 sQ[NPAD * DH];    // 80x32
  __shared__ _Float16 sK[NPAD * DH];    // 80x32
  __shared__ _Float16 sVt[DH * KPAD];   // 32x96 (v transposed: [dim][token])
  __shared__ float    sS[NPAD * NPAD];  // 80x80 scores (prefilled with bias)
  __shared__ _Float16 sP[NPAD * KPAD];  // 80x96 attn weights (f16, zero-padded)

  const int tid  = threadIdx.x;
  const int win  = blockIdx.x >> 3;
  const int h    = blockIdx.x & 7;
  const int wave = tid >> 5;
  const int lane = tid & 31;
  const int half = lane >> 4;
  const int lrow = lane & 15;
  const int mtiles = nPad >> 4;

  const size_t baseRow = (size_t)win * nTok;

  // stage q, k: 16B chunks, zero-filled padded rows
  for (int i = tid; i < nPad * 4; i += 256) {
    int t = i >> 2, c = i & 3;
    uint4 qv = make_uint4(0u, 0u, 0u, 0u);
    uint4 kv = make_uint4(0u, 0u, 0u, 0u);
    if (t < nTok) {
      const _Float16* row = QKV + (baseRow + t) * 768;
      qv = *(const uint4*)(row + h * DH + c * 8);
      kv = *(const uint4*)(row + 256 + h * DH + c * 8);
    }
    *(uint4*)(&sQ[t * DH + c * 8]) = qv;
    *(uint4*)(&sK[t * DH + c * 8]) = kv;
  }
  // stage v transposed: coalesced 16B global reads, scatter to sVt[d][t]
  for (int i = tid; i < nTok * 4; i += 256) {
    int t = i >> 2, c = i & 3;
    union { uint4 q; _Float16 e[8]; } u;
    u.q = *(const uint4*)(QKV + (baseRow + t) * 768 + 512 + h * DH + c * 8);
#pragma unroll
    for (int e = 0; e < 8; ++e) sVt[(c * 8 + e) * kPad + t] = u.e[e];
  }
  // zero v-transpose K padding (columns nTok..kPad-1)
  if (kPad > nTok) {
    int tail = kPad - nTok;
    for (int i = tid; i < DH * tail; i += 256) {
      int d = i / tail, t = nTok + i % tail;
      sVt[d * kPad + t] = (_Float16)0.f;
    }
  }
  // prefill scores with rel-pos bias (or zero); consumed as WMMA C operand
  for (int i = tid; i < nPad * nPad; i += 256) {
    int m = i / nPad, n = i % nPad;
    float v = 0.f;
    if (bias != nullptr && m < nTok && n < nTok)
      v = bias[((size_t)h * nTok + m) * nTok + n];
    sS[i] = v;
  }
  __syncthreads();

  // scores: sim = (q*s) @ k^T + bias, bias rides in as the C operand
  for (int tIdx = wave; tIdx < mtiles * mtiles; tIdx += 8) {
    int mt = tIdx / mtiles, nt = tIdx % mtiles;
    v16h a = ldsFrag(&sQ[(mt * 16 + lrow) * DH], half);
    v16h b = ldsFrag(&sK[(nt * 16 + lrow) * DH], half);
    float* cbase = &sS[(mt * 16 + half * 8) * nPad + nt * 16 + lrow];
    v8f c;
#pragma unroll
    for (int r = 0; r < 8; ++r) c[r] = cbase[r * nPad];  // batched ds loads
    c = wmma_f16(a, b, c);
#pragma unroll
    for (int r = 0; r < 8; ++r) cbase[r * nPad] = c[r];  // batched ds stores
  }
  __syncthreads();

  // row softmax -> f16 P (zero padding beyond nTok)
  for (int r = tid; r < nPad; r += 256) {
    if (r < nTok) {
      float mx = -3.0e38f;
      for (int j = 0; j < nTok; ++j) mx = fmaxf(mx, sS[r * nPad + j]);
      float sum = 0.f;
      for (int j = 0; j < nTok; ++j) {
        float e = __expf(sS[r * nPad + j] - mx);
        sS[r * nPad + j] = e;
        sum += e;
      }
      float inv = 1.f / sum;
      for (int j = 0; j < kPad; ++j)
        sP[r * kPad + j] = (j < nTok) ? (_Float16)(sS[r * nPad + j] * inv)
                                      : (_Float16)0.f;
    } else {
      for (int j = 0; j < kPad; ++j) sP[r * kPad + j] = (_Float16)0.f;
    }
  }
  __syncthreads();

  // O_h = P @ V  (M=nPad, N=32, K=kPad)
  for (int tIdx = wave; tIdx < mtiles * 2; tIdx += 8) {
    int mt = tIdx >> 1, nt2 = tIdx & 1;
    v8f acc = {};
    for (int kk = 0; kk < kPad; kk += 32) {
      v16h a = ldsFrag(&sP[(mt * 16 + lrow) * kPad + kk], half);
      v16h b = ldsFrag(&sVt[(nt2 * 16 + lrow) * kPad + kk], half);
      acc = wmma_f16(a, b, acc);
    }
#pragma unroll
    for (int r = 0; r < 8; ++r) {
      int m = mt * 16 + r + half * 8;
      int n = nt2 * 16 + lrow;
      if (m < nTok)
        O[(baseRow + m) * DIMM + h * DH + n] = (_Float16)acc[r];
    }
  }
}

// ---------------- LayerNorm: one wave per 256-dim token ----------------

__global__ __launch_bounds__(256) void k_ln(const float* __restrict__ X,
                                            const float* __restrict__ g,
                                            _Float16* __restrict__ H, int M) {
  int gwarp = (int)((blockIdx.x * 256 + threadIdx.x) >> 5);
  int lane  = threadIdx.x & 31;
  if (gwarp >= M) return;
  const float* row = X + (size_t)gwarp * DIMM;
  float v[8];
  float s = 0.f;
#pragma unroll
  for (int i = 0; i < 8; ++i) { v[i] = row[lane + i * 32]; s += v[i]; }
#pragma unroll
  for (int o = 16; o > 0; o >>= 1) s += __shfl_xor(s, o, 32);
  float mean = s * (1.f / 256.f);
  float s2 = 0.f;
#pragma unroll
  for (int i = 0; i < 8; ++i) { float d = v[i] - mean; s2 += d * d; }
#pragma unroll
  for (int o = 16; o > 0; o >>= 1) s2 += __shfl_xor(s2, o, 32);
  float rs = rsqrtf(s2 * (1.f / 256.f) + EPSV);
  _Float16* orow = H + (size_t)gwarp * DIMM;
#pragma unroll
  for (int i = 0; i < 8; ++i) {
    int c = lane + i * 32;
    orow[c] = (_Float16)((v[i] - mean) * rs * g[c]);
  }
}

// ---------------- Elementwise helpers ----------------

// dst[l][n][k] = (f16) src[l][k][n]  (pre-transpose weights for GEMM B)
__global__ void k_cvt_w(const float* __restrict__ src, _Float16* __restrict__ dst,
                        int K, int N, int layers) {
  size_t i = (size_t)blockIdx.x * 256 + threadIdx.x;
  size_t tot = (size_t)layers * N * K;
  if (i >= tot) return;
  int l = (int)(i / ((size_t)N * K));
  int rem = (int)(i % ((size_t)N * K));
  int n = rem / K, k = rem % K;
  dst[i] = (_Float16)src[(size_t)l * K * N + (size_t)k * N + n];
}

// expand rel-pos table (225, 8) -> (8, 65, 65) with zero row/col 0
__global__ void k_bias(const float* __restrict__ table, float* __restrict__ out) {
  int i = blockIdx.x * 256 + threadIdx.x;
  if (i >= HEADS * NTOK * NTOK) return;
  int h = i / (NTOK * NTOK);
  int rem = i % (NTOK * NTOK);
  int m = rem / NTOK, n = rem % NTOK;
  float v = 0.f;
  if (m > 0 && n > 0) {
    int a = m - 1, b = n - 1;
    int ax = a >> 3, ay = a & 7, bx = b >> 3, by = b & 7;
    int rx = ax - bx + 7, ry = ay - by + 7;
    v = table[(rx * 15 + ry) * HEADS + h];
  }
  out[i] = v;
}

// local (16,64,64,256) -> windowed X at token slots 1..64
__global__ void k_build_x(const float* __restrict__ local, float* __restrict__ X) {
  size_t i = (size_t)blockIdx.x * 256 + threadIdx.x;  // 16*64*64*256 elements
  int d = (int)(i & 255);
  size_t t = i >> 8;
  int xc = (int)(t & 63); t >>= 6;
  int y  = (int)(t & 63);
  int b  = (int)(t >> 6);
  int win = b * 64 + (y >> 3) * 8 + (xc >> 3);
  int tok = 1 + (y & 7) * 8 + (xc & 7);
  X[((size_t)win * NTOK + tok) * DIMM + d] = local[i];
}

__global__ void k_unwin(const float* __restrict__ X, float* __restrict__ out) {
  size_t i = (size_t)blockIdx.x * 256 + threadIdx.x;
  int d = (int)(i & 255);
  size_t t = i >> 8;
  int xc = (int)(t & 63); t >>= 6;
  int y  = (int)(t & 63);
  int b  = (int)(t >> 6);
  int win = b * 64 + (y >> 3) * 8 + (xc >> 3);
  int tok = 1 + (y & 7) * 8 + (xc & 7);
  out[i] = X[((size_t)win * NTOK + tok) * DIMM + d];
}

// X[win][0][:] = XR[win][:]
__global__ void k_scatter_region(const float* __restrict__ XR, float* __restrict__ X) {
  int i = blockIdx.x * 256 + threadIdx.x;  // 1024*256
  int w = i >> 8, d = i & 255;
  X[(size_t)w * NTOK * DIMM + d] = XR[i];
}

__global__ void k_gather_region(const float* __restrict__ X, float* __restrict__ XR) {
  int i = blockIdx.x * 256 + threadIdx.x;
  int w = i >> 8, d = i & 255;
  XR[i] = X[(size_t)w * NTOK * DIMM + d];
}

__global__ void k_copy_f32(const float* __restrict__ src, float* __restrict__ dst,
                           int n) {
  int i = blockIdx.x * 256 + threadIdx.x;
  if (i < n) dst[i] = src[i];
}

// ---------------------------------------------------------------------------

extern "C" void kernel_launch(void* const* d_in, const int* in_sizes, int n_in,
                              void* d_out, int out_size, void* d_ws, size_t ws_size,
                              hipStream_t stream) {
  (void)in_sizes; (void)n_in; (void)out_size; (void)ws_size;

  const float* local_in   = (const float*)d_in[0];
  const float* region_in  = (const float*)d_in[1];
  const float* bias_table = (const float*)d_in[2];
  const float* ln_attn_g  = (const float*)d_in[3];
  const float* w_qkv      = (const float*)d_in[4];
  const float* w_out      = (const float*)d_in[5];
  const float* b_out      = (const float*)d_in[6];
  const float* ln_mlp_g   = (const float*)d_in[7];
  const float* w1         = (const float*)d_in[8];
  const float* b1         = (const float*)d_in[9];
  const float* w2         = (const float*)d_in[10];
  const float* b2         = (const float*)d_in[11];
  float* outp = (float*)d_out;

  const int MW = NWIN * NTOK;  // 66560 window tokens (multiple of 128)
  const int MR = RWIN * RTOK;  // 1024 region tokens

  char* wsBase = (char*)d_ws;
  size_t off = 0;
  auto take = [&](size_t bytes) -> void* {
    void* p = wsBase + off;
    off = (off + bytes + 255) & ~(size_t)255;
    return p;
  };

  float*    X    = (float*)take((size_t)MW * DIMM * 4);      // windowed tokens f32
  float*    XR   = (float*)take((size_t)MR * DIMM * 4);      // region tokens f32
  _Float16* H    = (_Float16*)take((size_t)MW * DIMM * 2);   // LN output
  _Float16* QG   = (_Float16*)take((size_t)MW * 1024 * 2);   // QKV / MLP hidden
  _Float16* Obf  = (_Float16*)take((size_t)MW * DIMM * 2);   // attn output
  _Float16* HR   = (_Float16*)take((size_t)MR * DIMM * 2);
  _Float16* QR   = (_Float16*)take((size_t)MR * 768 * 2);
  _Float16* ORb  = (_Float16*)take((size_t)MR * DIMM * 2);
  _Float16* WqkvT = (_Float16*)take((size_t)4 * 768 * 256 * 2);
  _Float16* WoT   = (_Float16*)take((size_t)4 * 256 * 256 * 2);
  _Float16* W1T   = (_Float16*)take((size_t)4 * 1024 * 256 * 2);
  _Float16* W2T   = (_Float16*)take((size_t)4 * 256 * 1024 * 2);
  float*    biasB = (float*)take((size_t)HEADS * NTOK * NTOK * 4);

  // ---- one-time prep ----
  k_cvt_w<<<3072, 256, 0, stream>>>(w_qkv, WqkvT, 256, 768, 4);
  k_cvt_w<<<1024, 256, 0, stream>>>(w_out, WoT, 256, 256, 4);
  k_cvt_w<<<4096, 256, 0, stream>>>(w1, W1T, 256, 1024, 4);
  k_cvt_w<<<4096, 256, 0, stream>>>(w2, W2T, 1024, 256, 4);
  k_bias<<<(HEADS * NTOK * NTOK + 255) / 256, 256, 0, stream>>>(bias_table, biasB);
  k_build_x<<<65536, 256, 0, stream>>>(local_in, X);
  k_copy_f32<<<(MR * DIMM + 255) / 256, 256, 0, stream>>>(region_in, XR, MR * DIMM);

  for (int l = 0; l < 4; ++l) {
    const _Float16* wq  = WqkvT + (size_t)l * 768 * 256;
    const _Float16* wo  = WoT   + (size_t)l * 256 * 256;
    const _Float16* wm1 = W1T   + (size_t)l * 1024 * 256;
    const _Float16* wm2 = W2T   + (size_t)l * 256 * 1024;
    const float* gA  = ln_attn_g + l * 256;
    const float* gM  = ln_mlp_g + l * 256;
    const float* bo  = b_out + l * 256;
    const float* bb1 = b1 + l * 1024;
    const float* bb2 = b2 + l * 256;

    // ---- region-to-region attention (n=64, no bias) ----
    k_ln<<<MR / 8, 256, 0, stream>>>(XR, gA, HR, MR);
    k_gemm<0><<<dim3(MR / 128, 768 / 64), 256, 0, stream>>>(
        HR, wq, nullptr, nullptr, QR, MR, 768, 256);
    k_attn<<<RWIN * HEADS, 256, 0, stream>>>(QR, nullptr, ORb, RTOK, RTOK, RTOK);
    k_gemm<2><<<dim3(MR / 128, 256 / 64), 256, 0, stream>>>(
        ORb, wo, bo, XR, nullptr, MR, 256, 256);
    k_scatter_region<<<MR * DIMM / 256, 256, 0, stream>>>(XR, X);

    // ---- windowed attention (n=65, rel-pos bias) ----
    k_ln<<<MW / 8, 256, 0, stream>>>(X, gA, H, MW);
    k_gemm<0><<<dim3(MW / 128, 768 / 64), 256, 0, stream>>>(
        H, wq, nullptr, nullptr, QG, MW, 768, 256);
    k_attn<<<NWIN * HEADS, 256, 0, stream>>>(QG, biasB, Obf, NTOK, NPAD, KPAD);
    k_gemm<2><<<dim3(MW / 128, 256 / 64), 256, 0, stream>>>(
        Obf, wo, bo, X, nullptr, MW, 256, 256);

    // ---- MLP ----
    k_ln<<<MW / 8, 256, 0, stream>>>(X, gM, H, MW);
    k_gemm<1><<<dim3(MW / 128, 1024 / 64), 256, 0, stream>>>(
        H, wm1, bb1, nullptr, QG, MW, 1024, 256);
    k_gemm<2><<<dim3(MW / 128, 256 / 64), 256, 0, stream>>>(
        QG, wm2, bb2, X, nullptr, MW, 256, 1024);

    k_gather_region<<<MR * DIMM / 256, 256, 0, stream>>>(X, XR);
  }

  // ---- outputs: local (16,64,64,256) then region (16,8,8,256) ----
  k_unwin<<<65536, 256, 0, stream>>>(X, outp);
  k_copy_f32<<<(MR * DIMM + 255) / 256, 256, 0, stream>>>(
      XR, outp + (size_t)16 * 64 * 64 * 256, MR * DIMM);
}
